// HGCN_45260365365550
// MI455X (gfx1250) — matrix operand order repaired and checked by reading, not compiled, not run on hardware
//
#include <hip/hip_runtime.h>
#include <hip/hip_bf16.h>
#include <math.h>

#define BB   8
#define SS   1024
#define HH   512
#define HH2  1024
#define HH3  1536
#define OPN  6
#define FINN 7168
#define ROWS (BB*SS)       // 8192
#define OPROWS (BB*OPN)    // 48
#define SCHUNK 16          // agg split along S
#define SLEN  (SS/SCHUNK)  // 64

typedef __bf16 bf16t;
typedef __attribute__((ext_vector_type(2)))  __bf16 bf16x2;
typedef __attribute__((ext_vector_type(4)))  __bf16 bf16x4;
typedef __attribute__((ext_vector_type(16))) __bf16 v16bf;
typedef __attribute__((ext_vector_type(8)))  float  v8f;

// ---------------------------------------------------------------- utilities
// wave32 butterfly + one cross-wave combine; red8 must hold >= 8 floats.
__device__ __forceinline__ float block_reduce_sum(float v, float* red8) {
  #pragma unroll
  for (int o = 16; o > 0; o >>= 1) v += __shfl_xor(v, o, 32);
  const int tid = threadIdx.x;
  if ((tid & 31) == 0) red8[tid >> 5] = v;
  __syncthreads();
  float s = 0.0f;
  #pragma unroll
  for (int i = 0; i < 8; ++i) s += red8[i];
  __syncthreads();
  return s;
}

// ------------------------------------------------ bf16 WMMA tiled GEMM
// C[M,N] = op(A[M,K] @ B[K,N] + bias), batched via blockIdx.z strides.
// f32 global -> bf16 LDS staging (double buffered), f32 accumulate.
// FULL=true: M,N multiples of 128 (no bounds checks). K always mult of 32.
template<bool FULL>
__global__ __launch_bounds__(256)
void gemm_bf16_wmma(const float* __restrict__ A, const float* __restrict__ Bm,
                    float* __restrict__ C, const float* __restrict__ bias,
                    int M, int N, int K,
                    long long strideA, long long strideB, long long strideC,
                    int relu)
{
  const long long bz = blockIdx.z;
  A  += bz * strideA;
  Bm += bz * strideB;
  C  += bz * strideC;

  const int tid  = threadIdx.x;
  const int wid  = tid >> 5;
  const int lane = tid & 31;
  const int wrow = wid >> 2;       // 0..1  (64 rows each)
  const int wcol = wid & 3;        // 0..3  (32 cols each)
  const int half = lane >> 4;      // K-group select per ISA layout
  const int l15  = lane & 15;

  const int m_blk = blockIdx.y * 128;
  const int n_blk = blockIdx.x * 128;

  __shared__ bf16t As[2][128][40];   // [m][k], padded
  __shared__ bf16t Bs[2][128][40];   // [n][k] (transposed), padded

  // staging thread mapping
  const int ar = tid >> 3;           // A row within tile (+32 per i)
  const int ac = (tid & 7) * 4;      // A k-offset (float4)
  const int kp = tid >> 5;           // B k-pair index (+8 per i)
  const int bn = (tid & 31) * 4;     // B n-offset (float4)

  float4 aval[4], bv0[2], bv1[2];
  v8f acc[4][2] = {};

  auto loadG = [&](int k0) {
    #pragma unroll
    for (int i = 0; i < 4; ++i) {
      int r = ar + 32 * i;
      if (FULL || (m_blk + r) < M)
        aval[i] = *(const float4*)(A + (long long)(m_blk + r) * K + k0 + ac);
      else
        aval[i] = make_float4(0.f, 0.f, 0.f, 0.f);
    }
    #pragma unroll
    for (int i = 0; i < 2; ++i) {
      int k2 = 2 * (kp + 8 * i);     // even k row of the pair
      if (FULL || (n_blk + bn) < N) {
        bv0[i] = *(const float4*)(Bm + (long long)(k0 + k2)     * N + n_blk + bn);
        bv1[i] = *(const float4*)(Bm + (long long)(k0 + k2 + 1) * N + n_blk + bn);
      } else {
        bv0[i] = make_float4(0.f, 0.f, 0.f, 0.f);
        bv1[i] = make_float4(0.f, 0.f, 0.f, 0.f);
      }
    }
  };

  auto storeL = [&](int buf) {
    #pragma unroll
    for (int i = 0; i < 4; ++i) {
      bf16x4 p;
      p[0] = (bf16t)aval[i].x; p[1] = (bf16t)aval[i].y;
      p[2] = (bf16t)aval[i].z; p[3] = (bf16t)aval[i].w;
      *(bf16x4*)&As[buf][ar + 32 * i][ac] = p;
    }
    #pragma unroll
    for (int i = 0; i < 2; ++i) {
      int k2 = 2 * (kp + 8 * i);
      bf16x2 p0; p0[0] = (bf16t)bv0[i].x; p0[1] = (bf16t)bv1[i].x;
      bf16x2 p1; p1[0] = (bf16t)bv0[i].y; p1[1] = (bf16t)bv1[i].y;
      bf16x2 p2; p2[0] = (bf16t)bv0[i].z; p2[1] = (bf16t)bv1[i].z;
      bf16x2 p3; p3[0] = (bf16t)bv0[i].w; p3[1] = (bf16t)bv1[i].w;
      *(bf16x2*)&Bs[buf][bn + 0][k2] = p0;
      *(bf16x2*)&Bs[buf][bn + 1][k2] = p1;
      *(bf16x2*)&Bs[buf][bn + 2][k2] = p2;
      *(bf16x2*)&Bs[buf][bn + 3][k2] = p3;
    }
  };

  auto compute = [&](int buf) {
    // B fragments: lane n = l15; K quarters interleaved across lane halves (same as A)
    v16bf bfrag[2];
    #pragma unroll
    for (int nt = 0; nt < 2; ++nt) {
      int n = wcol * 32 + nt * 16 + l15;
      #pragma unroll
      for (int j = 0; j < 8; ++j) {
        int kk = ((j < 4) ? (2 * j) : (16 + 2 * (j - 4))) + 8 * half;
        bfrag[nt][2 * j]     = Bs[buf][n][kk];
        bfrag[nt][2 * j + 1] = Bs[buf][n][kk + 1];
      }
    }
    // A fragments: lane m = l15; K quarters interleaved across lane halves
    #pragma unroll
    for (int mt = 0; mt < 4; ++mt) {
      int mr = wrow * 64 + mt * 16 + l15;
      v16bf afrag;
      #pragma unroll
      for (int j = 0; j < 8; ++j) {
        int kb = ((j < 4) ? (2 * j) : (16 + 2 * (j - 4))) + 8 * half;
        afrag[2 * j]     = As[buf][mr][kb];
        afrag[2 * j + 1] = As[buf][mr][kb + 1];
      }
      #pragma unroll
      for (int nt = 0; nt < 2; ++nt) {
        acc[mt][nt] = __builtin_amdgcn_wmma_f32_16x16x32_bf16(
            false, afrag, false, bfrag[nt], (short)0, acc[mt][nt], false, false);
      }
    }
  };

  const int T = K >> 5;
  loadG(0);
  storeL(0);
  __syncthreads();

  for (int t = 0; t < T; ++t) {
    const int cur = t & 1;
    if (t + 1 < T) {
      loadG((t + 1) * 32);
      if (t + 2 < T) {   // hint the tile after next into cache
        __builtin_prefetch(A + (long long)(m_blk + ar) * K + (t + 2) * 32 + ac, 0, 1);
        __builtin_prefetch(Bm + (long long)((t + 2) * 32 + 2 * kp) * N + n_blk + bn, 0, 1);
      }
    }
    compute(cur);
    __syncthreads();
    if (t + 1 < T) storeL(1 - cur);
    __syncthreads();
  }

  // epilogue: C/D layout — lane<16: N=lane, M=v; lane>=16: N=lane-16, M=v+8
  #pragma unroll
  for (int mt = 0; mt < 4; ++mt) {
    #pragma unroll
    for (int nt = 0; nt < 2; ++nt) {
      int col = n_blk + wcol * 32 + nt * 16 + l15;
      if (FULL || col < N) {
        float bv = bias ? bias[col] : 0.0f;
        #pragma unroll
        for (int v = 0; v < 8; ++v) {
          int row = m_blk + wrow * 64 + mt * 16 + (half ? (v + 8) : v);
          if (FULL || row < M) {
            float val = acc[mt][nt][v] + bv;
            if (relu) val = fmaxf(val, 0.0f);
            C[(long long)row * N + col] = val;
          }
        }
      }
    }
  }
}

// ------------------------------------------------ elementwise builders
__global__ __launch_bounds__(256)
void build_wall_kernel(const float* __restrict__ attn, const float* __restrict__ svec,
                       const float* __restrict__ enc, float* __restrict__ w_all)
{
  long long idx = (long long)blockIdx.x * 256 + threadIdx.x;  // over ROWS*H2
  if (idx >= (long long)ROWS * HH2) return;
  int c   = (int)(idx & (HH2 - 1));
  long long row = idx >> 10;                 // b*S+s
  int b = (int)(row >> 10);
  float v;
  if (c < HH) v = attn[row] * svec[b * HH + c];
  else        v = enc[row * HH + (c - HH)];
  w_all[idx] = v;
}

__global__ __launch_bounds__(256)
void build_adj_kernel(const float* __restrict__ ww, const int* __restrict__ ex,
                      float* __restrict__ adj)
{
  long long idx = (long long)blockIdx.x * 256 + threadIdx.x;  // B*S*S
  if (idx >= (long long)BB * SS * SS) return;
  adj[idx] = (ex[idx] == 1) ? ww[idx] : 0.0f;
}

__global__ __launch_bounds__(256)
void zero_kernel(float* __restrict__ p, int n)
{
  int idx = blockIdx.x * 256 + threadIdx.x;
  if (idx < n) p[idx] = 0.0f;
}

// ------------------------------------------------ non-standard LayerNorm
// std = sqrt(sum((x-mean)^2))  (no 1/N), optional residual (row % res_mod)
__global__ __launch_bounds__(256)
void ln1_kernel(const float* __restrict__ x, const float* __restrict__ a,
                const float* __restrict__ bvec, const float* __restrict__ resid,
                int res_mod, int len, float* __restrict__ out)
{
  __shared__ float red[8];
  const int row = blockIdx.x;
  const int tid = threadIdx.x;
  const int nper = len >> 8;                 // len in {512,1024}
  const float* xr = x + (long long)row * len;
  float lx[4];
  float s = 0.0f;
  for (int i = 0; i < nper; ++i) { lx[i] = xr[tid + 256 * i]; s += lx[i]; }
  float mean = block_reduce_sum(s, red) / (float)len;
  float ss = 0.0f;
  for (int i = 0; i < nper; ++i) { float d = lx[i] - mean; ss += d * d; }
  float stdv = sqrtf(block_reduce_sum(ss, red)) + 1e-6f;
  const float* rr = resid ? (resid + (long long)(row % res_mod) * len) : nullptr;
  for (int i = 0; i < nper; ++i) {
    int c = tid + 256 * i;
    float v = a[c] * (lx[i] - mean) / stdv + bvec[c];
    if (rr) v += rr[c];
    out[(long long)row * len + c] = v;
  }
}

// ------------------------------------------------ L2-normalize rows
__global__ __launch_bounds__(256)
void l2norm_kernel(const float* __restrict__ x, float* __restrict__ out, int len)
{
  __shared__ float red[8];
  const int row = blockIdx.x;
  const int tid = threadIdx.x;
  const int nper = len >> 8;
  const float* xr = x + (long long)row * len;
  float lx[4], ss = 0.0f;
  for (int i = 0; i < nper; ++i) { lx[i] = xr[tid + 256 * i]; ss += lx[i] * lx[i]; }
  float nrm = sqrtf(block_reduce_sum(ss, red)) + 1e-30f;
  for (int i = 0; i < nper; ++i)
    out[(long long)row * len + tid + 256 * i] = lx[i] / nrm;
}

// ------------------------------------------------ word->op adjacency (normalized)
__global__ __launch_bounds__(256)
void woadj_kernel(const float* __restrict__ word_op, const int* __restrict__ seq_mask,
                  float* __restrict__ wo_adjN)
{
  __shared__ float red[8];
  const int bo = blockIdx.x;           // b*OPN+o
  const int b = bo / OPN, o = bo % OPN;
  const int tid = threadIdx.x;
  float s = 0.0f;
  for (int si = tid; si < SS; si += 256) {
    float v = seq_mask[b * SS + si] ? 0.0f
            : word_op[((long long)b * SS + si) * OPN + o];
    s += v;
  }
  float denom = block_reduce_sum(s, red) + 1e-30f;
  for (int si = tid; si < SS; si += 256) {
    float v = seq_mask[b * SS + si] ? 0.0f
            : word_op[((long long)b * SS + si) * OPN + o];
    wo_adjN[(long long)bo * SS + si] = v / denom;
  }
}

// ------------------------------------------------ agg[b,o,:] += wo_adjN[b,o,s-chunk] @ w2w[b,s-chunk,:]
// grid (BB, SCHUNK): split along S for memory-level parallelism; combine with atomics.
__global__ __launch_bounds__(256)
void agg_kernel(const float* __restrict__ wo_adjN, const float* __restrict__ w2w,
                float* __restrict__ agg)
{
  __shared__ float wsh[OPN * SLEN];
  const int b  = blockIdx.x;
  const int s0 = blockIdx.y * SLEN;
  const int tid = threadIdx.x;
  for (int i = tid; i < OPN * SLEN; i += 256) {
    int o = i / SLEN, si = i % SLEN;
    wsh[i] = wo_adjN[((long long)b * OPN + o) * SS + s0 + si];
  }
  __syncthreads();

  float acc[OPN][4] = {};
  for (int si = 0; si < SLEN; ++si) {
    float wv[OPN];
    #pragma unroll
    for (int o = 0; o < OPN; ++o) wv[o] = wsh[o * SLEN + si];
    #pragma unroll
    for (int i = 0; i < 4; ++i) {
      float x = w2w[((long long)b * SS + s0 + si) * HH2 + tid + 256 * i];
      #pragma unroll
      for (int o = 0; o < OPN; ++o) acc[o][i] += wv[o] * x;
    }
  }
  #pragma unroll
  for (int o = 0; o < OPN; ++o)
    #pragma unroll
    for (int i = 0; i < 4; ++i)
      atomicAdd(&agg[((long long)b * OPN + o) * HH2 + tid + 256 * i], acc[o][i]);
}

// ------------------------------------------------ op_all = [op_trans, ops]
__global__ __launch_bounds__(256)
void build_opall_kernel(const float* __restrict__ op_trans, const float* __restrict__ ops,
                        float* __restrict__ op_all)
{
  int idx = blockIdx.x * 256 + threadIdx.x;   // OPROWS*HH3
  if (idx >= OPROWS * HH3) return;
  int r = idx / HH3, c = idx % HH3;
  int o = r % OPN;
  float v = (c < HH) ? op_trans[r * HH + c] : ops[o * HH2 + (c - HH)];
  op_all[idx] = v;
}

// ------------------------------------------------ per-batch forget/rel offsets
// off[b][j] = sum_{o,c} on[b,o,c] * W[(H2 + o*H2 + c)*OPN + j]
__global__ __launch_bounds__(256)
void op_offsets_kernel(const float* __restrict__ on, const float* __restrict__ fw,
                       const float* __restrict__ rw,
                       float* __restrict__ off_f, float* __restrict__ off_r)
{
  __shared__ float red[8];
  const int b = blockIdx.x;
  const int tid = threadIdx.x;
  float pf[OPN] = {}, pr[OPN] = {};
  for (int i = tid; i < OPN * HH2; i += 256) {
    float v = on[(long long)b * OPN * HH2 + i];
    long long rowW = HH2 + i;
    #pragma unroll
    for (int j = 0; j < OPN; ++j) {
      pf[j] += v * fw[rowW * OPN + j];
      pr[j] += v * rw[rowW * OPN + j];
    }
  }
  for (int j = 0; j < OPN; ++j) {
    float t = block_reduce_sum(pf[j], red);
    if (tid == 0) off_f[b * OPN + j] = t;
    t = block_reduce_sum(pr[j], red);
    if (tid == 0) off_r[b * OPN + j] = t;
  }
}

// ------------------------------------------------ fused per-row update_graph
__global__ __launch_bounds__(256)
void update_kernel(const float* __restrict__ w2w,
                   const float* __restrict__ fw, const float* __restrict__ fb,
                   const float* __restrict__ rw, const float* __restrict__ rb,
                   const float* __restrict__ uw, const float* __restrict__ ub,
                   const float* __restrict__ off_f, const float* __restrict__ off_r,
                   const float* __restrict__ word_op, float* __restrict__ out_wo)
{
  __shared__ float red[8];
  __shared__ float dots_f[OPN], dots_r[OPN];
  const int row = blockIdx.x;          // b*S+s
  const int b = row >> 10;
  const int tid = threadIdx.x;
  const float* xr = w2w + (long long)row * HH2;
  float lx[4], ss = 0.0f;
  #pragma unroll
  for (int i = 0; i < 4; ++i) { lx[i] = xr[tid + 256 * i]; ss += lx[i] * lx[i]; }
  float nrm = sqrtf(block_reduce_sum(ss, red)) + 1e-30f;
  float wn[4];
  #pragma unroll
  for (int i = 0; i < 4; ++i) wn[i] = lx[i] / nrm;

  for (int j = 0; j < OPN; ++j) {
    float pf = 0.0f, pr = 0.0f;
    #pragma unroll
    for (int i = 0; i < 4; ++i) {
      int c = tid + 256 * i;
      pf += wn[i] * fw[(long long)c * OPN + j];
      pr += wn[i] * rw[(long long)c * OPN + j];
    }
    float tf = block_reduce_sum(pf, red);
    float tr = block_reduce_sum(pr, red);
    if (tid == 0) { dots_f[j] = tf; dots_r[j] = tr; }
  }
  __syncthreads();

  if (tid == 0) {
    float wop[OPN], rel[OPN], forget[OPN], logits[OPN];
    #pragma unroll
    for (int k = 0; k < OPN; ++k)
      wop[k] = word_op[(long long)row * OPN + k];
    #pragma unroll
    for (int j = 0; j < OPN; ++j) {
      float fj = dots_f[j] + off_f[b * OPN + j] + fb[j];
      forget[j] = fmaxf(1.0f / (1.0f + expf(-fj)), 0.99f);
      rel[j] = dots_r[j] + off_r[b * OPN + j] + rb[j];
    }
    #pragma unroll
    for (int j = 0; j < OPN; ++j) {
      float l = ub[j];
      #pragma unroll
      for (int k = 0; k < OPN; ++k) {
        l += wop[k] * uw[k * OPN + j];
        l += rel[k] * uw[(OPN + k) * OPN + j];
      }
      logits[j] = fmaxf(l, 0.0f);
    }
    float mx = logits[0];
    #pragma unroll
    for (int j = 1; j < OPN; ++j) mx = fmaxf(mx, logits[j]);
    float es[OPN], sum = 0.0f;
    #pragma unroll
    for (int j = 0; j < OPN; ++j) { es[j] = expf(logits[j] - mx); sum += es[j]; }
    #pragma unroll
    for (int j = 0; j < OPN; ++j) {
      float upd = es[j] / sum;
      out_wo[(long long)row * OPN + j] = forget[j] * wop[j] + (1.0f - forget[j]) * upd;
    }
  }
}

// ---------------------------------------------------------------- launcher
extern "C" void kernel_launch(void* const* d_in, const int* in_sizes, int n_in,
                              void* d_out, int out_size, void* d_ws, size_t ws_size,
                              hipStream_t stream) {
  (void)in_sizes; (void)n_in; (void)out_size; (void)ws_size;
  const float* enc      = (const float*)d_in[0];
  const float* ops      = (const float*)d_in[1];
  const float* svec     = (const float*)d_in[2];
  const float* attn     = (const float*)d_in[3];
  const float* ww       = (const float*)d_in[4];
  const float* word_op  = (const float*)d_in[5];
  const int*   exmat    = (const int*)d_in[6];
  const int*   seq_mask = (const int*)d_in[7];
  const float* gc1_w    = (const float*)d_in[8];
  const float* gc1_b    = (const float*)d_in[9];
  const float* gc2_w    = (const float*)d_in[10];
  const float* gc2_b    = (const float*)d_in[11];
  const float* norm_a   = (const float*)d_in[12];
  const float* norm_b   = (const float*)d_in[13];
  const float* norm1_a  = (const float*)d_in[14];
  const float* norm1_b  = (const float*)d_in[15];
  const float* norm2_a  = (const float*)d_in[16];
  const float* norm2_b  = (const float*)d_in[17];
  const float* wo_w     = (const float*)d_in[18];
  const float* wo_b     = (const float*)d_in[19];
  const float* otrans_w = (const float*)d_in[20];
  const float* otrans_b = (const float*)d_in[21];
  const float* forget_w = (const float*)d_in[22];
  const float* forget_b = (const float*)d_in[23];
  const float* rel_w    = (const float*)d_in[24];
  const float* rel_b    = (const float*)d_in[25];
  const float* update_w = (const float*)d_in[26];
  const float* update_b = (const float*)d_in[27];

  float* out = (float*)d_out;
  float* out_op_o = out;                       // (B,OP,H2) = 49152
  float* out_wo   = out + OPROWS * HH2;        // (B,S,OP)  = 49152

  // workspace layout (floats)
  float* ws = (float*)d_ws;
  float* w_all    = ws;                        size_t off = (size_t)ROWS * HH2;
  float* adj      = ws + off;                  off += (size_t)BB * SS * SS;
  float* xw       = ws + off;                  off += (size_t)ROWS * HH;
  float* h        = ws + off;                  off += (size_t)ROWS * HH;
  float* hg       = ws + off;                  off += (size_t)ROWS * HH2;
  float* w2w      = ws + off;                  off += (size_t)ROWS * HH2;
  float* wo_adjN  = ws + off;                  off += (size_t)OPROWS * SS;
  float* agg      = ws + off;                  off += (size_t)OPROWS * HH2;
  float* op_trans = ws + off;                  off += (size_t)OPROWS * HH;
  float* op_all   = ws + off;                  off += (size_t)OPROWS * HH3;
  float* op_h     = ws + off;                  off += (size_t)OPROWS * HH2;
  float* onrm     = ws + off;                  off += (size_t)OPROWS * HH2;
  float* off_f    = ws + off;                  off += (size_t)OPROWS;
  float* off_r    = ws + off;                  off += (size_t)OPROWS;

  // 1. w_all = [attn^T * s, enc];  adj = mask(word_word)
  {
    long long n = (long long)ROWS * HH2;
    build_wall_kernel<<<dim3((unsigned)((n + 255) / 256)), 256, 0, stream>>>(attn, svec, enc, w_all);
    long long na = (long long)BB * SS * SS;
    build_adj_kernel<<<dim3((unsigned)((na + 255) / 256)), 256, 0, stream>>>(ww, exmat, adj);
  }

  // 2. GCN layer 1: xw = w_all @ gc1_w ; h = relu(adj @ xw + gc1_b)
  gemm_bf16_wmma<true><<<dim3(HH / 128, ROWS / 128, 1), 256, 0, stream>>>(
      w_all, gc1_w, xw, nullptr, ROWS, HH, HH2, 0, 0, 0, 0);
  gemm_bf16_wmma<true><<<dim3(HH / 128, SS / 128, BB), 256, 0, stream>>>(
      adj, xw, h, gc1_b, SS, HH, SS,
      (long long)SS * SS, (long long)SS * HH, (long long)SS * HH, 1);

  // 3. GCN layer 2: hg = h @ gc2_w ; w2w_raw = adj @ hg + gc2_b
  gemm_bf16_wmma<true><<<dim3(HH2 / 128, ROWS / 128, 1), 256, 0, stream>>>(
      h, gc2_w, hg, nullptr, ROWS, HH2, HH, 0, 0, 0, 0);
  gemm_bf16_wmma<true><<<dim3(HH2 / 128, SS / 128, BB), 256, 0, stream>>>(
      adj, hg, w2w, gc2_b, SS, HH2, SS,
      (long long)SS * SS, (long long)SS * HH2, (long long)SS * HH2, 0);

  // 4. w2w = ln1(w2w_raw) + w_all  (in place)
  ln1_kernel<<<ROWS, 256, 0, stream>>>(w2w, norm_a, norm_b, w_all, ROWS, HH2, w2w);

  // 5. word->op aggregation (split-S partials + atomics)
  woadj_kernel<<<OPROWS, 256, 0, stream>>>(word_op, seq_mask, wo_adjN);
  zero_kernel<<<(OPROWS * HH2 + 255) / 256, 256, 0, stream>>>(agg, OPROWS * HH2);
  agg_kernel<<<dim3(BB, SCHUNK), 256, 0, stream>>>(wo_adjN, w2w, agg);

  // 6. op_trans = ln1(relu(agg @ wo_w + wo_b))
  gemm_bf16_wmma<false><<<dim3(HH / 128, 1, 1), 256, 0, stream>>>(
      agg, wo_w, op_trans, wo_b, OPROWS, HH, HH2, 0, 0, 0, 1);
  ln1_kernel<<<OPROWS, 256, 0, stream>>>(op_trans, norm1_a, norm1_b, nullptr, 1, HH, op_trans);

  // 7. op_o = ln1(relu([op_trans, ops] @ otrans_w + otrans_b)) + ops  -> d_out
  build_opall_kernel<<<(OPROWS * HH3 + 255) / 256, 256, 0, stream>>>(op_trans, ops, op_all);
  gemm_bf16_wmma<false><<<dim3(HH2 / 128, 1, 1), 256, 0, stream>>>(
      op_all, otrans_w, op_h, otrans_b, OPROWS, HH2, HH3, 0, 0, 0, 1);
  ln1_kernel<<<OPROWS, 256, 0, stream>>>(op_h, norm2_a, norm2_b, ops, OPN, HH2, out_op_o);

  // 8. update_graph: on = normalize(op_o); per-batch offsets; fused per-row update
  l2norm_kernel<<<OPROWS, 256, 0, stream>>>(out_op_o, onrm, HH2);
  op_offsets_kernel<<<BB, 256, 0, stream>>>(onrm, forget_w, rel_w, off_f, off_r);
  update_kernel<<<ROWS, 256, 0, stream>>>(w2w, forget_w, forget_b, rel_w, rel_b,
                                          update_w, update_b, off_f, off_r,
                                          word_op, out_wo);
}
